// BiGAT_graphcl_34574486733593
// MI455X (gfx1250) — matrix operand order repaired and checked by reading, not compiled, not run on hardware
//
#include <hip/hip_runtime.h>
#include <hip/hip_bf16.h>
#include <math.h>

// ---------------- problem constants (match reference) ----------------
#define NN    131072           // nodes
#define EE    131072           // edges
#define ETOT  (EE + NN)        // edges + self loops
#define BB    128              // graphs
#define INF_  768
#define HIDD  256
#define OUTD  256
#define HH    4
#define CCH   64               // per-head channels (both layers)
#define CD    1024             // (OUT+HID)*2
#define NC    4

typedef __attribute__((ext_vector_type(16))) __bf16 v16bf;
typedef __attribute__((ext_vector_type(8)))  __bf16 v8bf;
typedef __attribute__((ext_vector_type(8)))  float  v8f;

// ---------------- small utility kernels ----------------
__global__ void k_fill_f32(float* p, size_t n, float v) {
    size_t i = (size_t)blockIdx.x * blockDim.x + threadIdx.x;
    if (i < n) p[i] = v;
}
__global__ void k_fill_i32(int* p, int n, int v) {
    int i = blockIdx.x * blockDim.x + threadIdx.x;
    if (i < n) p[i] = v;
}
__global__ void k_first_min(const int* __restrict__ batch, int* __restrict__ first) {
    int i = blockIdx.x * blockDim.x + threadIdx.x;
    if (i < NN) atomicMin(&first[batch[i]], i);
}
__global__ void k_first_clamp(int* first) {
    int g = blockIdx.x * blockDim.x + threadIdx.x;
    if (g < BB) { int v = first[g]; if (v > NN - 1) v = NN - 1; if (v < 0) v = 0; first[g] = v; }
}

// ---- one-time weight transpose+convert: Bt[n][k] (bf16) = Bm[k][n] (f32) ----
__global__ void k_conv_bt(const float* __restrict__ Bm, __bf16* __restrict__ Bt,
                          int K, int Ncol)
{
    int i = blockIdx.x * blockDim.x + threadIdx.x;   // n*K + k
    if (i >= K * Ncol) return;
    int n = i / K, k = i - n * K;
    Bt[i] = (__bf16)Bm[(size_t)k * Ncol + n];
}

// ---------------- bf16 WMMA GEMM: C[M,N] = A[M,K] @ B[K,N] (+bias, relu) ----
// B is pre-transposed bf16: Bt[n*ldBt + k].  ALIGNMENT CONTRACT:
// grid.y = M/128 (M multiple of 128), K % 32 == 0, Ncol % 64 == 0 via grid.x,
// A rows & Bt rows 16-byte aligned.
// block = 256 threads = 8 waves. Block tile 128x64; each wave 16x64 (4 WMMAs/K-step).
// Software pipelined: global loads for tile k+1 overlap WMMAs of tile k.
__global__ void k_gemm_wmma(const float* __restrict__ A, const __bf16* __restrict__ Bt,
                            const float* __restrict__ bias, float* __restrict__ C,
                            int K, int ldA, int ldBt, int ldC, int relu)
{
    __shared__ __bf16 lA[8][16][32];   // per-wave A tile (16x32), 8 KB
    __shared__ __bf16 lBt[64][32];     // B tile transposed [n][k], 4 KB
    const int lane = threadIdx.x & 31;
    const int wave = threadIdx.x >> 5;
    const int n0   = blockIdx.x * 64;
    const int m0   = blockIdx.y * 128 + wave * 16;
    const int half = lane >> 4;        // 0: lanes 0-15, 1: lanes 16-31
    const int mrow = lane & 15;
    // A fill mapping: lane -> (row, 16-col group)
    const int ar = lane >> 1;
    const int ac = (lane & 1) * 16;
    // B fill mapping: thread -> (col n, 8-elem k group)
    const int bn = threadIdx.x & 63;
    const int bk = (threadIdx.x >> 6) * 8;

    const float4* aptr = (const float4*)(A + (size_t)(m0 + ar) * ldA + ac);
    const v8bf*   bptr = (const v8bf*)(Bt + (size_t)(n0 + bn) * ldBt + bk);

    v8f acc0 = {}, acc1 = {}, acc2 = {}, acc3 = {};
    // prime the pipeline: tile k0 = 0
    float4 qa0 = aptr[0], qa1 = aptr[1], qa2 = aptr[2], qa3 = aptr[3];
    v8bf   qb  = bptr[0];

    for (int k0 = 0; k0 < K; k0 += 32) {
        // ---- convert + stage current tile into LDS ----
        {
            v8bf t0, t1;
            t0[0] = (__bf16)qa0.x; t0[1] = (__bf16)qa0.y; t0[2] = (__bf16)qa0.z; t0[3] = (__bf16)qa0.w;
            t0[4] = (__bf16)qa1.x; t0[5] = (__bf16)qa1.y; t0[6] = (__bf16)qa1.z; t0[7] = (__bf16)qa1.w;
            t1[0] = (__bf16)qa2.x; t1[1] = (__bf16)qa2.y; t1[2] = (__bf16)qa2.z; t1[3] = (__bf16)qa2.w;
            t1[4] = (__bf16)qa3.x; t1[5] = (__bf16)qa3.y; t1[6] = (__bf16)qa3.z; t1[7] = (__bf16)qa3.w;
            *(v8bf*)&lA[wave][ar][ac]     = t0;
            *(v8bf*)&lA[wave][ar][ac + 8] = t1;
            *(v8bf*)&lBt[bn][bk]          = qb;
        }
        __syncthreads();
        // ---- issue global loads for the NEXT tile (overlap with WMMAs) ----
        if (k0 + 32 < K) {
            const float4* an = (const float4*)((const float*)aptr + ldA * 0 + (k0 + 32) - k0);
            // recompute cleanly from base each step:
            an = (const float4*)(A + (size_t)(m0 + ar) * ldA + (k0 + 32) + ac);
            qa0 = an[0]; qa1 = an[1]; qa2 = an[2]; qa3 = an[3];
            qb  = *(const v8bf*)(Bt + (size_t)(n0 + bn) * ldBt + (k0 + 32) + bk);
        }
        // ---- A fragment: lane half selects K {0..15} vs {16..31}
        v8bf alo = *(v8bf*)&lA[wave][mrow][half * 16];
        v8bf ahi = *(v8bf*)&lA[wave][mrow][half * 16 + 8];
        v16bf af = __builtin_shufflevector(alo, ahi, 0, 1, 2, 3, 4, 5, 6, 7,
                                           8, 9, 10, 11, 12, 13, 14, 15);
        // ---- B fragments (LDS rows contiguous in K) + 4 WMMAs
#pragma unroll
        for (int nt = 0; nt < 4; ++nt) {
            v8bf b0 = *(v8bf*)&lBt[nt * 16 + mrow][half * 16];
            v8bf b1 = *(v8bf*)&lBt[nt * 16 + mrow][half * 16 + 8];
            v16bf bf = __builtin_shufflevector(b0, b1, 0, 1, 2, 3, 4, 5, 6, 7,
                                               8, 9, 10, 11, 12, 13, 14, 15);
            if (nt == 0) acc0 = __builtin_amdgcn_wmma_f32_16x16x32_bf16(false, af, false, bf, (short)0, acc0, false, false);
            if (nt == 1) acc1 = __builtin_amdgcn_wmma_f32_16x16x32_bf16(false, af, false, bf, (short)0, acc1, false, false);
            if (nt == 2) acc2 = __builtin_amdgcn_wmma_f32_16x16x32_bf16(false, af, false, bf, (short)0, acc2, false, false);
            if (nt == 3) acc3 = __builtin_amdgcn_wmma_f32_16x16x32_bf16(false, af, false, bf, (short)0, acc3, false, false);
        }
        __syncthreads();
    }
    // ---- store: C/D layout VGPR v -> M = v + 8*half, N = lane&15
#pragma unroll
    for (int nt = 0; nt < 4; ++nt) {
        const v8f* accp = (nt == 0) ? &acc0 : (nt == 1) ? &acc1 : (nt == 2) ? &acc2 : &acc3;
        int n = n0 + nt * 16 + mrow;
        float bv = bias ? bias[n] : 0.f;
#pragma unroll
        for (int v = 0; v < 8; ++v) {
            int m = m0 + v + half * 8;
            float o = (*accp)[v] + bv;
            if (relu) o = fmaxf(o, 0.f);
            C[(size_t)m * ldC + n] = o;
        }
    }
}

// ---------------- GAT attention pieces ----------------
__global__ void k_attn_scores(const float* __restrict__ h, const float* __restrict__ a_s,
                              const float* __restrict__ a_d, float* __restrict__ es,
                              float* __restrict__ ed)
{
    int i = blockIdx.x * blockDim.x + threadIdx.x;       // node*H + head
    if (i >= NN * HH) return;
    int node = i >> 2, hd = i & 3;
    const float* hp = h + (size_t)node * HIDD + hd * CCH;
    const float* as = a_s + hd * CCH;
    const float* ad = a_d + hd * CCH;
    float s = 0.f, d = 0.f;
#pragma unroll 8
    for (int c = 0; c < CCH; ++c) { float v = hp[c]; s += v * as[c]; d += v * ad[c]; }
    es[i] = s; ed[i] = d;
}

__device__ __forceinline__ void atomicMaxF(float* addr, float v) {
    if (v >= 0.f) atomicMax((int*)addr, __float_as_int(v));
    else          atomicMin((unsigned int*)addr, __float_as_uint(v));
}

__global__ void k_edge_max(const int* __restrict__ src, const int* __restrict__ dst,
                           const float* __restrict__ es, const float* __restrict__ ed,
                           float* __restrict__ ebuf, float* __restrict__ mx)
{
    int i = blockIdx.x * blockDim.x + threadIdx.x;       // edge*H + head
    if (i >= ETOT * HH) return;
    int ei = i >> 2, hd = i & 3;
    int s = (ei < EE) ? src[ei] : (ei - EE);
    int d = (ei < EE) ? dst[ei] : (ei - EE);
    float e = es[s * HH + hd] + ed[d * HH + hd];
    e = (e > 0.f) ? e : 0.2f * e;                        // leaky_relu(0.2)
    ebuf[i] = e;
    atomicMaxF(&mx[d * HH + hd], e);
}

__global__ void k_edge_exp(const int* __restrict__ src, const int* __restrict__ dst,
                           float* __restrict__ ebuf, const float* __restrict__ mx,
                           float* __restrict__ denom)
{
    int i = blockIdx.x * blockDim.x + threadIdx.x;
    if (i >= ETOT * HH) return;
    int ei = i >> 2, hd = i & 3;
    int d = (ei < EE) ? dst[ei] : (ei - EE);
    float p = expf(ebuf[i] - mx[d * HH + hd]);
    ebuf[i] = p;                                         // in-place: e -> p
    atomicAdd(&denom[d * HH + hd], p);
}

__global__ void k_edge_aggr(const int* __restrict__ src, const int* __restrict__ dst,
                            const float* __restrict__ pbuf, const float* __restrict__ denom,
                            const float* __restrict__ hfeat, float* __restrict__ agg)
{
    int i = blockIdx.x * blockDim.x + threadIdx.x;
    if (i >= ETOT * HH) return;
    int ei = i >> 2, hd = i & 3;
    int s = (ei < EE) ? src[ei] : (ei - EE);
    int d = (ei < EE) ? dst[ei] : (ei - EE);
    float w = pbuf[i] / denom[d * HH + hd];
    const float* hs = hfeat + (size_t)s * HIDD + hd * CCH;
    float* od = agg + (size_t)d * HIDD + hd * CCH;
#pragma unroll 8
    for (int c = 0; c < CCH; ++c) atomicAdd(&od[c], hs[c] * w);
}

// ---------------- (bias) -> BN -> LN -> (relu), one wave per row ----------
template <int D>
__global__ void k_ln_transform(float* __restrict__ buf, const float* __restrict__ bias,
                               const float* __restrict__ bn_g, const float* __restrict__ bn_b,
                               const float* __restrict__ ln_g, const float* __restrict__ ln_b,
                               int M, int relu)
{
    int row = blockIdx.x * (blockDim.x >> 5) + (threadIdx.x >> 5);
    if (row >= M) return;
    const int lane = threadIdx.x & 31;
    const int per = D / 32;
    float vals[D / 32];
    float* r = buf + (size_t)row * D;
    float sum = 0.f;
#pragma unroll
    for (int j = 0; j < per; ++j) {
        int c = lane * per + j;
        float v = r[c];
        if (bias) v += bias[c];
        if (bn_g) v = v * bn_g[c] + bn_b[c];
        vals[j] = v; sum += v;
    }
    for (int m = 16; m; m >>= 1) sum += __shfl_xor(sum, m, 32);
    float mean = sum / (float)D;
    float var = 0.f;
#pragma unroll
    for (int j = 0; j < per; ++j) { float t = vals[j] - mean; var += t * t; }
    for (int m = 16; m; m >>= 1) var += __shfl_xor(var, m, 32);
    float inv = rsqrtf(var / (float)D + 1e-5f);
#pragma unroll
    for (int j = 0; j < per; ++j) {
        int c = lane * per + j;
        float v = (vals[j] - mean) * inv * ln_g[c] + ln_b[c];
        if (relu) v = fmaxf(v, 0.f);
        r[c] = v;
    }
}

// ---------------- root handling ----------------
__global__ void k_gather_root(const float* __restrict__ x, const int* __restrict__ first,
                              float* __restrict__ Aroot)
{
    int i = blockIdx.x * blockDim.x + threadIdx.x;       // g*IN + c
    if (i >= BB * INF_) return;
    int g = i / INF_, c = i % INF_;
    Aroot[i] = x[(size_t)first[g] * INF_ + c];
}
__global__ void k_add_root(float* __restrict__ h2, const float* __restrict__ rootT,
                           const int* __restrict__ batch)
{
    size_t i = (size_t)blockIdx.x * blockDim.x + threadIdx.x;  // n*256 + c
    if (i >= (size_t)NN * OUTD) return;
    int n = (int)(i >> 8), c = (int)(i & 255);
    h2[i] += rootT[batch[n] * OUTD + c];
}

// ---------------- pooling (scatter mean) ----------------
__global__ void k_pool_sum(const float* __restrict__ xf, const int* __restrict__ batch,
                           float* __restrict__ psum, float* __restrict__ pcnt)
{
    int nd = blockIdx.x;            // node
    int c = threadIdx.x;            // channel 0..255
    int g = batch[nd];
    atomicAdd(&psum[g * OUTD + c], xf[(size_t)nd * OUTD + c]);
    if (c == 0) atomicAdd(&pcnt[g], 1.0f);
}
__global__ void k_pool_finish(const float* __restrict__ psum, const float* __restrict__ pcnt,
                              const float* __restrict__ x2, const int* __restrict__ first,
                              float* __restrict__ cbuf, int coff)
{
    int i = blockIdx.x * blockDim.x + threadIdx.x;       // g*512 + c
    if (i >= BB * 512) return;
    int g = i >> 9, c = i & 511;
    float v;
    if (c < OUTD) v = psum[g * OUTD + c] / fmaxf(pcnt[g], 1.0f);
    else          v = x2[(size_t)first[g] * HIDD + (c - OUTD)];
    cbuf[g * CD + coff + c] = v;
}

// ---------------- head ----------------
__global__ void k_head_att(const float* __restrict__ a1, const float* __restrict__ A2,
                           const float* __restrict__ a2b, float* __restrict__ att)
{
    int g = blockIdx.x * blockDim.x + threadIdx.x;
    if (g >= BB) return;
    float l0 = a2b[0], l1 = a2b[1];
    const float* r = a1 + g * 256;
#pragma unroll 8
    for (int c = 0; c < 256; ++c) { l0 += r[c] * A2[c * 2]; l1 += r[c] * A2[c * 2 + 1]; }
    float m = fmaxf(l0, l1);
    float e0 = expf(l0 - m), e1 = expf(l1 - m);
    float s = e0 + e1;
    att[g * 2] = e0 / s; att[g * 2 + 1] = e1 / s;
}

__global__ void k_head_final(const float* __restrict__ cbuf, const float* __restrict__ att,
                             const float* __restrict__ bn_g, const float* __restrict__ bn_b,
                             const float* __restrict__ ln_g, const float* __restrict__ ln_b,
                             const float* __restrict__ fcW, const float* __restrict__ fcb,
                             float* __restrict__ out)
{
    int g = blockIdx.x * (blockDim.x >> 5) + (threadIdx.x >> 5);
    if (g >= BB) return;
    const int lane = threadIdx.x & 31;
    float a0 = att[g * 2], a1v = att[g * 2 + 1];
    const float* row = cbuf + (size_t)g * CD;
    float vals[32]; float sum = 0.f;
#pragma unroll
    for (int j = 0; j < 32; ++j) {
        int c = lane * 32 + j;
        float v = row[c] * ((c < 512) ? a0 : a1v);
        v = v * bn_g[c] + bn_b[c];
        vals[j] = v; sum += v;
    }
    for (int m = 16; m; m >>= 1) sum += __shfl_xor(sum, m, 32);
    float mean = sum / (float)CD;
    float var = 0.f;
#pragma unroll
    for (int j = 0; j < 32; ++j) { float t = vals[j] - mean; var += t * t; }
    for (int m = 16; m; m >>= 1) var += __shfl_xor(var, m, 32);
    float inv = rsqrtf(var / (float)CD + 1e-5f);
    float lg[NC] = {0.f, 0.f, 0.f, 0.f};
#pragma unroll
    for (int j = 0; j < 32; ++j) {
        int c = lane * 32 + j;
        float v = (vals[j] - mean) * inv * ln_g[c] + ln_b[c];
#pragma unroll
        for (int k = 0; k < NC; ++k) lg[k] += v * fcW[c * NC + k];
    }
#pragma unroll
    for (int k = 0; k < NC; ++k)
        for (int m = 16; m; m >>= 1) lg[k] += __shfl_xor(lg[k], m, 32);
    if (lane == 0) {
        float z[NC], mz = -1e30f;
#pragma unroll
        for (int k = 0; k < NC; ++k) { z[k] = lg[k] + fcb[k]; mz = fmaxf(mz, z[k]); }
        float se = 0.f;
#pragma unroll
        for (int k = 0; k < NC; ++k) se += expf(z[k] - mz);
        float lse = mz + logf(se);
#pragma unroll
        for (int k = 0; k < NC; ++k) out[g * NC + k] = z[k] - lse;
    }
}

// ================= orchestration =================
static inline unsigned cdiv(size_t a, unsigned b) { return (unsigned)((a + b - 1) / b); }

extern "C" void kernel_launch(void* const* d_in, const int* in_sizes, int n_in,
                              void* d_out, int out_size, void* d_ws, size_t ws_size,
                              hipStream_t stream)
{
    (void)in_sizes; (void)n_in; (void)out_size; (void)ws_size;
    const float* x    = (const float*)d_in[0];
    const int*   eidx = (const int*)d_in[1];      // [2,E] row-major
    const int*   batch = (const int*)d_in[2];
    auto P = [&](int i) { return (const float*)d_in[i]; };
    const int HP = 35;  // param bases: td=3, bu=19, hp=35

    // ---- carve workspace ----
    char* w = (char*)d_ws; size_t off = 0;
    auto carve = [&](size_t bytes) -> char* {
        char* p = w + off; off += (bytes + 255) & ~(size_t)255; return p;
    };
    int*    first = (int*)   carve((size_t)BB * 4);
    float*  bufA  = (float*) carve((size_t)NN * HIDD * 4);   // h1 / h2
    float*  bufB  = (float*) carve((size_t)NN * HIDD * 4);   // agg1 -> x2
    float*  bufC  = (float*) carve((size_t)NN * OUTD * 4);   // agg2 -> x_final
    float*  es    = (float*) carve((size_t)NN * HH * 4);
    float*  edv   = (float*) carve((size_t)NN * HH * 4);
    float*  mx    = (float*) carve((size_t)NN * HH * 4);
    float*  den   = (float*) carve((size_t)NN * HH * 4);
    float*  ebuf  = (float*) carve((size_t)ETOT * HH * 4);   // e -> p in place
    float*  Aroot = (float*) carve((size_t)BB * INF_ * 4);
    float*  rootT = (float*) carve((size_t)BB * OUTD * 4);
    float*  psum  = (float*) carve((size_t)BB * OUTD * 4);
    float*  pcnt  = (float*) carve((size_t)BB * 4);
    float*  cbuf  = (float*) carve((size_t)BB * CD * 4);
    float*  a1buf = (float*) carve((size_t)BB * 256 * 4);
    float*  att   = (float*) carve((size_t)BB * 2 * 4);
    __bf16* BtW1  = (__bf16*)carve((size_t)HIDD * INF_ * 2); // [256][768]
    __bf16* BtW2  = (__bf16*)carve((size_t)OUTD * CD / 1024 * 1024 * 2); // [256][1024]
    __bf16* BtA1  = (__bf16*)carve((size_t)256 * CD * 2);    // [256][1024]
    float*  outf  = (float*)d_out;

    // ---- root indices (shared by both branches) ----
    k_fill_i32<<<cdiv(BB, 256), 256, 0, stream>>>(first, BB, 0x7fffffff);
    k_first_min<<<cdiv(NN, 256), 256, 0, stream>>>(batch, first);
    k_first_clamp<<<1, 256, 0, stream>>>(first);
    k_gather_root<<<cdiv((size_t)BB * INF_, 256), 256, 0, stream>>>(x, first, Aroot);

    // ---- two branches: TD (src,dst) -> cols 512..1023, BU (dst,src) -> cols 0..511
    struct Cfg { const int* s; const int* d; int pb; int co; };
    Cfg cfg[2] = { { eidx,      eidx + EE, 3,  512 },     // TD
                   { eidx + EE, eidx,      19, 0   } };   // BU
    const unsigned EG = cdiv((size_t)ETOT * HH, 256);
    for (int b = 0; b < 2; ++b) {
        const int pb = cfg[b].pb;
        const int* S = cfg[b].s; const int* D = cfg[b].d;
        // one-time weight transpose/convert for this branch
        k_conv_bt<<<cdiv((size_t)INF_ * HIDD, 256), 256, 0, stream>>>(P(pb + 0), BtW1, INF_, HIDD);
        k_conv_bt<<<cdiv((size_t)CD   * OUTD, 256), 256, 0, stream>>>(P(pb + 8), BtW2, CD, OUTD);
        // GAT layer 1: h1 = x @ W1 (WMMA), 131072x768x256
        k_gemm_wmma<<<dim3(HIDD / 64, NN / 128), 256, 0, stream>>>(
            x, BtW1, nullptr, bufA, INF_, INF_, INF_, HIDD, 0);
        k_attn_scores<<<cdiv((size_t)NN * HH, 256), 256, 0, stream>>>(bufA, P(pb + 1), P(pb + 2), es, edv);
        k_fill_f32<<<cdiv((size_t)NN * HH, 256), 256, 0, stream>>>(mx,  (size_t)NN * HH, -INFINITY);
        k_fill_f32<<<cdiv((size_t)NN * HH, 256), 256, 0, stream>>>(den, (size_t)NN * HH, 0.f);
        k_edge_max<<<EG, 256, 0, stream>>>(S, D, es, edv, ebuf, mx);
        k_edge_exp<<<EG, 256, 0, stream>>>(S, D, ebuf, mx, den);
        k_fill_f32<<<cdiv((size_t)NN * HIDD, 256), 256, 0, stream>>>(bufB, (size_t)NN * HIDD, 0.f);
        k_edge_aggr<<<EG, 256, 0, stream>>>(S, D, ebuf, den, bufA, bufB);
        // +b1, BN1, LN1, ReLU  -> x2 in bufB
        k_ln_transform<HIDD><<<cdiv(NN, 8), 256, 0, stream>>>(
            bufB, P(pb + 3), P(pb + 4), P(pb + 5), P(pb + 6), P(pb + 7), NN, 1);
        // layer-2 split GEMMs: rootT = Aroot @ W2[256:,:] (k offset 256 in Bt rows),
        //                      h2    = x2    @ W2[:256,:]
        k_gemm_wmma<<<dim3(OUTD / 64, BB / 128), 256, 0, stream>>>(
            Aroot, BtW2 + 256, nullptr, rootT, INF_, INF_, CD, OUTD, 0);
        k_gemm_wmma<<<dim3(OUTD / 64, NN / 128), 256, 0, stream>>>(
            bufB, BtW2, nullptr, bufA, HIDD, HIDD, CD, OUTD, 0);
        k_add_root<<<cdiv((size_t)NN * OUTD, 256), 256, 0, stream>>>(bufA, rootT, batch);
        // GAT layer 2 edge softmax + aggregation
        k_attn_scores<<<cdiv((size_t)NN * HH, 256), 256, 0, stream>>>(bufA, P(pb + 9), P(pb + 10), es, edv);
        k_fill_f32<<<cdiv((size_t)NN * HH, 256), 256, 0, stream>>>(mx,  (size_t)NN * HH, -INFINITY);
        k_fill_f32<<<cdiv((size_t)NN * HH, 256), 256, 0, stream>>>(den, (size_t)NN * HH, 0.f);
        k_edge_max<<<EG, 256, 0, stream>>>(S, D, es, edv, ebuf, mx);
        k_edge_exp<<<EG, 256, 0, stream>>>(S, D, ebuf, mx, den);
        k_fill_f32<<<cdiv((size_t)NN * OUTD, 256), 256, 0, stream>>>(bufC, (size_t)NN * OUTD, 0.f);
        k_edge_aggr<<<EG, 256, 0, stream>>>(S, D, ebuf, den, bufA, bufC);
        // +b2, BN2, LN2, ReLU -> x_final in bufC
        k_ln_transform<OUTD><<<cdiv(NN, 8), 256, 0, stream>>>(
            bufC, P(pb + 11), P(pb + 12), P(pb + 13), P(pb + 14), P(pb + 15), NN, 1);
        // scatter-mean pooling + root-extend -> cbuf columns [co, co+512)
        k_fill_f32<<<cdiv((size_t)BB * OUTD, 256), 256, 0, stream>>>(psum, (size_t)BB * OUTD, 0.f);
        k_fill_f32<<<1, 256, 0, stream>>>(pcnt, (size_t)BB, 0.f);
        k_pool_sum<<<NN, OUTD, 0, stream>>>(bufC, batch, psum, pcnt);
        k_pool_finish<<<cdiv((size_t)BB * 512, 256), 256, 0, stream>>>(psum, pcnt, bufB, first, cbuf, cfg[b].co);
    }

    // ---- head ----
    // a1 = relu(LN(c @ A1 + a1b))  (WMMA GEMM, bias fused), 128x1024x256
    k_conv_bt<<<cdiv((size_t)CD * 256, 256), 256, 0, stream>>>(P(HP + 0), BtA1, CD, 256);
    k_gemm_wmma<<<dim3(256 / 64, BB / 128), 256, 0, stream>>>(
        cbuf, BtA1, P(HP + 1), a1buf, CD, CD, CD, 256, 0);
    k_ln_transform<256><<<cdiv(BB, 8), 256, 0, stream>>>(
        a1buf, nullptr, nullptr, nullptr, P(HP + 2), P(HP + 3), BB, 1);
    k_head_att<<<1, BB, 0, stream>>>(a1buf, P(HP + 4), P(HP + 5), att);
    k_head_final<<<cdiv(BB, 8), 256, 0, stream>>>(
        cbuf, att, P(HP + 6), P(HP + 7), P(HP + 8), P(HP + 9), P(HP + 10), P(HP + 11), outf);
}
